// BiMamba4KT_78194174591437
// MI455X (gfx1250) — matrix operand order repaired and checked by reading, not compiled, not run on hardware
//
#include <hip/hip_runtime.h>
#include <hip/hip_bf16.h>

// ---------------- problem constants ----------------
#define BQ    64
#define LQ    200
#define DQ    256
#define DSQ   16
#define DIQ   512
#define DTRQ  16
#define KCONV 4
#define MQ    (BQ*LQ)      // 12800 rows (= 100 * 128)
#define NQUES 3162
#define NQUESP 3200        // padded to 64

typedef __attribute__((ext_vector_type(16))) _Float16 v16h;
typedef __attribute__((ext_vector_type(8)))  _Float16 v8h;
typedef __attribute__((ext_vector_type(8)))  float    v8f;
typedef __attribute__((ext_vector_type(4)))  float    v4f;
typedef __attribute__((ext_vector_type(4)))  unsigned v4u;

#define USE_ASYNC_B 1

__device__ __forceinline__ float sigmoidf_(float x){ return 1.f/(1.f+__expf(-x)); }

// ACT: 0=none, 1=gelu(tanh approx), 2=softplus
template<int ACT>
__device__ __forceinline__ float act_f(float x){
  if (ACT==1){
    float x3 = x*x*x;
    return 0.5f*x*(1.f + tanhf(0.7978845608f*(x + 0.044715f*x3)));
  }
  if (ACT==2){
    return (x > 20.f) ? x : log1pf(__expf(x));
  }
  return x;
}

// ---------- weight prep: fp32 [K,N] -> f16 K-pair-interleaved dwords [Kp/2, Np] ----------
// dword(kp,n) = { f16(src[2kp+1,n]) << 16 | f16(src[2kp,n]) }, zero padded.
__global__ void __launch_bounds__(256)
prep_w_kernel(const float* __restrict__ src, int K, int N,
              unsigned* __restrict__ dst, int Kp2, int Np)
{
  int idx = blockIdx.x*256 + threadIdx.x;
  if (idx >= Kp2*Np) return;
  int n = idx % Np, kp = idx / Np;
  int k0 = 2*kp, k1 = k0 + 1;
  float lo = (k0 < K && n < N) ? src[k0*N + n] : 0.f;
  float hi = (k1 < K && n < N) ? src[k1*N + n] : 0.f;
  union { _Float16 h; unsigned short s; } ul, uh;
  ul.h = (_Float16)lo; uh.h = (_Float16)hi;
  dst[idx] = ((unsigned)uh.s << 16) | (unsigned)ul.s;
}

// ---------------- WMMA GEMM: C = act(A[M,K] @ B + bias) ----------------
// A fp32 (converted to f16 in LDS), B pre-converted padded f16 dwords.
// Tile 128(M) x 64(N) x 32(K); 8 waves, each wave: 16 rows x 64 cols = 4 accumulators.
// Double-buffered LDS; B tile staged by gfx1250 async-to-LDS DMA (ASYNCcnt),
// overlapped with WMMA compute of the current tile.
#define BM 128
#define BN 64
#define BK 32
#define AS_STRIDE 40   // halfs; 80 B rows (16B aligned, conflict-free fragment reads)
#define BS_STRIDE 68   // dwords; 272 B rows (16B aligned)

__device__ __forceinline__ void stage_b_async(const unsigned* gb, unsigned* ldsdst){
  unsigned ldsoff = (unsigned)(unsigned long long)ldsdst;
  asm volatile("global_load_async_to_lds_b128 %0, %1, off"
               :: "v"(ldsoff), "v"((unsigned long long)gb) : "memory");
}

__device__ __forceinline__ void cvt_store_a(_Float16* dst, v4f a0, v4f a1, v4f a2, v4f a3){
  v8h h0, h1;
  h0[0]=(_Float16)a0[0]; h0[1]=(_Float16)a0[1]; h0[2]=(_Float16)a0[2]; h0[3]=(_Float16)a0[3];
  h0[4]=(_Float16)a1[0]; h0[5]=(_Float16)a1[1]; h0[6]=(_Float16)a1[2]; h0[7]=(_Float16)a1[3];
  h1[0]=(_Float16)a2[0]; h1[1]=(_Float16)a2[1]; h1[2]=(_Float16)a2[2]; h1[3]=(_Float16)a2[3];
  h1[4]=(_Float16)a3[0]; h1[5]=(_Float16)a3[1]; h1[6]=(_Float16)a3[2]; h1[7]=(_Float16)a3[3];
  *(v8h*)dst       = h0;
  *(v8h*)(dst + 8) = h1;
}

template<int ACT>
__global__ void __launch_bounds__(256)
gemm_wmma(const float* __restrict__ A, int lda,
          const unsigned* __restrict__ Bq, int ldbq,  // dwords per kp-row (= Np)
          const float* __restrict__ bias,
          float* __restrict__ C, int ldc,
          int M, int N, int K)
{
  __shared__ _Float16 As[2][BM][AS_STRIDE];
  __shared__ unsigned BsD[2][BK/2][BS_STRIDE];

  const int tid  = threadIdx.x;
  const int lane = tid & 31;
  const int wid  = tid >> 5;          // 0..7 -> 16-row slab
  const int half = lane >> 4;
  const int l15  = lane & 15;
  const int bm0  = blockIdx.y * BM;
  const int bn0  = blockIdx.x * BN;

  // A staging: thread -> row ar (0..127), 16 consecutive floats at ac0
  const int ar  = tid >> 1;
  const int ac0 = (tid & 1) << 4;
  // B staging: thread -> kp (0..15), 4 dwords at bc0
  const int bkp = tid >> 4;
  const int bc0 = (tid & 15) << 2;

  v8f acc[4] = {};
  const int nk = K / BK;
  v4f a0, a1, a2, a3;

  // ---- prologue: stage tile 0 ----
  {
    const unsigned* gb = Bq + (size_t)bkp*ldbq + bn0 + bc0;
#if USE_ASYNC_B
    stage_b_async(gb, &BsD[0][bkp][bc0]);
#else
    *(v4u*)&BsD[0][bkp][bc0] = *(const v4u*)gb;
#endif
    const float* ap = A + (size_t)(bm0 + ar)*lda + ac0;
    a0 = *(const v4f*)(ap + 0);  a1 = *(const v4f*)(ap + 4);
    a2 = *(const v4f*)(ap + 8);  a3 = *(const v4f*)(ap + 12);
    cvt_store_a(&As[0][ar][ac0], a0, a1, a2, a3);
#if USE_ASYNC_B
    asm volatile("s_wait_asynccnt 0x0" ::: "memory");
#endif
    __syncthreads();
  }

  for (int kt = 0; kt < nk; ++kt) {
    const int  cur  = kt & 1;
    const int  nxt  = cur ^ 1;
    const bool more = (kt + 1) < nk;

    // ---- prefetch tile kt+1 (async DMA for B, global loads for A) ----
    if (more) {
      const unsigned* gb = Bq + (size_t)((kt+1)*(BK/2) + bkp)*ldbq + bn0 + bc0;
#if USE_ASYNC_B
      stage_b_async(gb, &BsD[nxt][bkp][bc0]);
#else
      *(v4u*)&BsD[nxt][bkp][bc0] = *(const v4u*)gb;
#endif
      const float* ap = A + (size_t)(bm0 + ar)*lda + (kt+1)*BK + ac0;
      a0 = *(const v4f*)(ap + 0);  a1 = *(const v4f*)(ap + 4);
      a2 = *(const v4f*)(ap + 8);  a3 = *(const v4f*)(ap + 12);
    }

    // ---- compute tile kt from the current buffers ----
    const int mrow = wid*16 + l15;
    union { v16h h; unsigned u[8]; } fa;
    #pragma unroll
    for (int v = 0; v < 8; ++v) {
      int kk = ((v >> 2) << 4) + (half << 3) + ((v & 3) << 1);
      fa.u[v] = *(const unsigned*)&As[cur][mrow][kk];
    }
    #pragma unroll
    for (int f = 0; f < 4; ++f) {
      union { v16h h; unsigned u[8]; } fb;
      #pragma unroll
      for (int v = 0; v < 8; ++v)
        fb.u[v] = BsD[cur][((v >> 2) << 3) + (half << 2) + (v & 3)][f*16 + l15];
      acc[f] = __builtin_amdgcn_wmma_f32_16x16x32_f16(false, fa.h, false, fb.h,
                                                      (short)0, acc[f], false, false);
    }

    // ---- finish staging tile kt+1, single barrier per iteration ----
    if (more) {
      cvt_store_a(&As[nxt][ar][ac0], a0, a1, a2, a3);
#if USE_ASYNC_B
      asm volatile("s_wait_asynccnt 0x0" ::: "memory");
#endif
      __syncthreads();
    }
  }

  // ---- epilogue (only place with N guards) ----
  const int mrow0 = bm0 + wid*16 + (half << 3);
  #pragma unroll
  for (int f = 0; f < 4; ++f) {
    int ncol = bn0 + f*16 + l15;
    if (ncol < N) {
      float bv = bias ? bias[ncol] : 0.f;
      #pragma unroll
      for (int r = 0; r < 8; ++r)
        C[(size_t)(mrow0 + r)*ldc + ncol] = act_f<ACT>(acc[f][r] + bv);
    }
  }
}

// ---------------- embeddings ----------------
__global__ void __launch_bounds__(256)
embed_kernel(const int* __restrict__ q, const int* __restrict__ qa,
             const float* __restrict__ diff, const float* __restrict__ qtab,
             const float* __restrict__ qatab,
             float* __restrict__ QA, float* __restrict__ QD)
{
  int idx = blockIdx.x*256 + threadIdx.x;
  if (idx >= MQ*DQ) return;
  int dd = idx & (DQ-1);
  int bl = idx >> 8;
  int qi = q[bl], qai = qa[bl];
  QA[idx] = qatab[qai*DQ + dd];
  QD[idx] = (1.f - diff[qi]) * qtab[qi*DQ + dd];
}

// ---------------- causal depthwise conv + SiLU (logical time = rev?flip:id) ----
__global__ void __launch_bounds__(256)
conv_silu_kernel(const float* __restrict__ XZ, const float* __restrict__ w,
                 const float* __restrict__ cb, float* __restrict__ XH, int rev)
{
  int idx = blockIdx.x*256 + threadIdx.x;   // over M*DI
  if (idx >= MQ*DIQ) return;
  int c  = idx & (DIQ-1);
  int bl = idx >> 9;
  int t  = bl % LQ;
  int b  = bl / LQ;
  float acc = cb[c];
  #pragma unroll
  for (int k = 0; k < KCONV; ++k) {
    int tt = t - (KCONV-1) + k;
    if (tt >= 0) {
      int pt = rev ? (LQ-1-tt) : tt;
      acc += XZ[(b*LQ + pt)*(2*DIQ) + c] * w[k*DIQ + c];
    }
  }
  XH[idx] = acc * sigmoidf_(acc);
}

// ---------------- selective scan (+ fused silu(z) gating); XDBL stride 64 ----
__global__ void __launch_bounds__(256)
scan_kernel(const float* __restrict__ XH, const float* __restrict__ DT,
            const float* __restrict__ XDBL, const float* __restrict__ A_log,
            const float* __restrict__ Dp, const float* __restrict__ XZ,
            float* __restrict__ Y, int rev)
{
  int b     = blockIdx.x >> 1;
  int chunk = blockIdx.x & 1;
  int ch    = chunk*256 + threadIdx.x;
  __shared__ float Bs[DSQ], Cs[DSQ];
  float A[DSQ], h[DSQ];
  float dp = Dp[ch];
  #pragma unroll
  for (int s = 0; s < DSQ; ++s) { A[s] = -__expf(A_log[ch*DSQ + s]); h[s] = 0.f; }
  for (int t = 0; t < LQ; ++t) {
    int row = b*LQ + t;
    if (threadIdx.x < DSQ)            Bs[threadIdx.x]       = XDBL[row*64 + DTRQ + threadIdx.x];
    else if (threadIdx.x < 2*DSQ)     Cs[threadIdx.x - DSQ] = XDBL[row*64 + DTRQ + DSQ + (threadIdx.x - DSQ)];
    __syncthreads();
    float u  = XH[row*DIQ + ch];
    float d  = DT[row*DIQ + ch];
    float du = d*u;
    float y  = 0.f;
    #pragma unroll
    for (int s = 0; s < DSQ; ++s) {
      h[s] = __expf(d*A[s])*h[s] + du*Bs[s];
      y += h[s]*Cs[s];
    }
    y += u*dp;
    int pt = rev ? (LQ-1-t) : t;
    float z = XZ[(b*LQ + pt)*(2*DIQ) + DIQ + ch];
    Y[row*DIQ + ch] = y * (z * sigmoidf_(z));
    __syncthreads();
  }
}

// ---------------- fwd/bwd accumulation with time remap ----------------
__global__ void __launch_bounds__(256)
bi_accum_kernel(const float* __restrict__ S, float* __restrict__ Dst, int rev, int first)
{
  int idx = blockIdx.x*256 + threadIdx.x;
  if (idx >= MQ*DQ) return;
  int dch = idx & (DQ-1);
  int bl  = idx >> 8;
  int t = bl % LQ, b = bl / LQ;
  int pt = rev ? (LQ-1-t) : t;
  float v = S[idx];
  float* d = Dst + (b*LQ + pt)*DQ + dch;
  if (first) *d = v; else *d += v;
}

// ---------------- residual add + LayerNorm over D=256 ----------------
__global__ void __launch_bounds__(256)
add_ln_kernel(const float* __restrict__ X1, const float* __restrict__ X2,
              const float* __restrict__ g, const float* __restrict__ be,
              float* __restrict__ O)
{
  int row = blockIdx.x;
  int j   = threadIdx.x;
  float x = X1[row*DQ + j] + X2[row*DQ + j];
  __shared__ float red[256];
  red[j] = x; __syncthreads();
  for (int s = 128; s > 0; s >>= 1) { if (j < s) red[j] += red[j+s]; __syncthreads(); }
  float mean = red[0] * (1.f/DQ); __syncthreads();
  float dv = x - mean;
  red[j] = dv*dv; __syncthreads();
  for (int s = 128; s > 0; s >>= 1) { if (j < s) red[j] += red[j+s]; __syncthreads(); }
  float var = red[0] * (1.f/DQ);
  O[row*DQ + j] = dv * rsqrtf(var + 1e-12f) * g[j] + be[j];
}

// ---------------- GRU recurrence (x-projections precomputed in XG) ----------
__global__ void __launch_bounds__(256)
gru_scan_kernel(const float* __restrict__ XG, const float* __restrict__ Uz,
                const float* __restrict__ Ur, const float* __restrict__ Un,
                float* __restrict__ HS)
{
  int b = blockIdx.x;
  int j = threadIdx.x;
  __shared__ float hsh[DQ];
  hsh[j] = 0.f;
  __syncthreads();
  for (int t = 0; t < LQ; ++t) {
    const float* xg = XG + (b*LQ + t)*768;
    float hz = 0.f, hr = 0.f, hn = 0.f;
    #pragma unroll 4
    for (int i = 0; i < DQ; ++i) {
      float hi = hsh[i];
      hz += hi * Uz[i*DQ + j];
      hr += hi * Ur[i*DQ + j];
      hn += hi * Un[i*DQ + j];
    }
    float z = sigmoidf_(xg[j]        + hz);
    float r = sigmoidf_(xg[DQ + j]   + hr);
    float n = tanhf   (xg[2*DQ + j] + r*hn);
    float hnew = (1.f - z)*hsh[j] + z*n;
    __syncthreads();
    hsh[j] = hnew;
    HS[(b*LQ + t)*DQ + j] = hnew;
    __syncthreads();
  }
}

// ---------------- final mix: a1*h + a2*d + a3*gru ----------------
__global__ void __launch_bounds__(256)
mix_kernel(const float* __restrict__ oh, const float* __restrict__ od,
           const float* __restrict__ gr, const float* __restrict__ a1,
           const float* __restrict__ a2, const float* __restrict__ a3,
           float* __restrict__ z0)
{
  int idx = blockIdx.x*256 + threadIdx.x;
  if (idx >= MQ*DQ) return;
  z0[idx] = a1[0]*oh[idx] + a2[0]*od[idx] + a3[0]*gr[idx];
}

// ---------------- host-side dispatch helpers ----------------
static inline void launch_gemm(hipStream_t st, const float* A, int lda,
                               const unsigned* Bq, const float* bias,
                               float* C, int ldc, int N, int Np, int K, int act)
{
  dim3 grid(Np/BN, MQ/BM), blk(256);
  if (act == 1)      gemm_wmma<1><<<grid, blk, 0, st>>>(A, lda, Bq, Np, bias, C, ldc, MQ, N, K);
  else if (act == 2) gemm_wmma<2><<<grid, blk, 0, st>>>(A, lda, Bq, Np, bias, C, ldc, MQ, N, K);
  else               gemm_wmma<0><<<grid, blk, 0, st>>>(A, lda, Bq, Np, bias, C, ldc, MQ, N, K);
}

static inline void prep_w(hipStream_t st, const float* src, int K, int N,
                          unsigned* dst, int Kp2, int Np)
{
  int tot = Kp2*Np;
  prep_w_kernel<<<(tot + 255)/256, 256, 0, st>>>(src, K, N, dst, Kp2, Np);
}

extern "C" void kernel_launch(void* const* d_in, const int* in_sizes, int n_in,
                              void* d_out, int out_size, void* d_ws, size_t ws_size,
                              hipStream_t stream)
{
  (void)in_sizes; (void)n_in; (void)out_size; (void)ws_size;
  // -------- flattened input order from setup_inputs() --------
  const int*   qa   = (const int*)  d_in[0];
  const int*   q    = (const int*)  d_in[1];
  const float* diff = (const float*)d_in[2];
  const float* qtab = (const float*)d_in[3];
  const float* qatab= (const float*)d_in[4];
  const float* Wz = (const float*)d_in[5],  *Uz = (const float*)d_in[6],  *bz = (const float*)d_in[7];
  const float* Wr = (const float*)d_in[8],  *Ur = (const float*)d_in[9],  *br = (const float*)d_in[10];
  const float* Wn = (const float*)d_in[11], *Un = (const float*)d_in[12], *bn = (const float*)d_in[13];
  const float* Wo = (const float*)d_in[14], *bo = (const float*)d_in[15];
  const int LBASE = 42;                           // only layer 1 matters (loop overwrites)
  const float* a1   = (const float*)d_in[68];
  const float* a2   = (const float*)d_in[69];
  const float* a3   = (const float*)d_in[70];
  const float* mixw = (const float*)d_in[71];
  const float* mixb = (const float*)d_in[72];
  const float* fcw  = (const float*)d_in[73];
  const float* fcb  = (const float*)d_in[74];
  float* out = (float*)d_out;

  // -------- workspace layout (floats) --------
  float* ws   = (float*)d_ws;
  float* QA   = ws;                    // [M,256]
  float* QD   = QA   + 3276800;
  float* OUTH = QD   + 3276800;
  float* OUTD = OUTH + 3276800;
  float* GRUO = OUTD + 3276800;
  float* BIG0 = GRUO + 3276800;        // [M,1024] (XZ / FFN-hidden)
  float* XH   = BIG0 + 13107200;       // [M,512]
  float* DTb  = XH   + 6553600;        // [M,512]
  float* Yb   = DTb  + 6553600;        // [M,512]
  float* XDBL = Yb   + 6553600;        // [M,64] padded
  float* TMPA = XDBL + 819200;         // [M,256]
  float* BI   = TMPA + 3276800;
  float* XG   = BI   + 3276800;        // [M,768]
  float* HS   = XG   + 9830400;
  float* Hb   = HS   + 3276800;
  // -------- f16 interleaved weight region (dwords) --------
  unsigned* WQ = (unsigned*)(Hb + 3276800);
  unsigned* wWin[2] = { WQ,            WQ + 131072 };            // 128 x 1024
  unsigned* wWx [2] = { WQ + 262144,   WQ + 278528 };            // 256 x 64
  unsigned* wWdt[2] = { WQ + 294912,   WQ + 303104 };            // 16  x 512
  unsigned* wWou[2] = { WQ + 311296,   WQ + 376832 };            // 256 x 256
  unsigned* wF1     = WQ + 442368;                               // 128 x 1024
  unsigned* wF2     = WQ + 573440;                               // 512 x 256
  unsigned* wGz     = WQ + 704512;                               // 128 x 256
  unsigned* wGr     = WQ + 737280;
  unsigned* wGn     = WQ + 770048;
  unsigned* wGo     = WQ + 802816;
  unsigned* wMix    = WQ + 835584;
  unsigned* wFc     = WQ + 868352;                               // 128 x 3200

  const int EW_BLOCKS = (MQ*DQ)/256;   // 12800

  // 0) weight prep (fp32 -> padded interleaved f16)
  for (int dir = 0; dir < 2; ++dir) {
    const int pb = LBASE + (dir ? 9 : 0);
    prep_w(stream, (const float*)d_in[pb+0], 256, 1024, wWin[dir], 128, 1024); // W_in
    prep_w(stream, (const float*)d_in[pb+3], 512,   48, wWx [dir], 256,   64); // W_x
    prep_w(stream, (const float*)d_in[pb+4],  16,  512, wWdt[dir],  16,  512); // W_dt (K 16->32)
    prep_w(stream, (const float*)d_in[pb+8], 512,  256, wWou[dir], 256,  256); // W_out
  }
  prep_w(stream, (const float*)d_in[LBASE+20], 256, 1024, wF1, 128, 1024);     // ffn w1
  prep_w(stream, (const float*)d_in[LBASE+22], 1024, 256, wF2, 512,  256);     // ffn w2
  prep_w(stream, Wz,   256, 256, wGz, 128, 256);
  prep_w(stream, Wr,   256, 256, wGr, 128, 256);
  prep_w(stream, Wn,   256, 256, wGn, 128, 256);
  prep_w(stream, Wo,   256, 256, wGo, 128, 256);
  prep_w(stream, mixw, 256, 256, wMix, 128, 256);
  prep_w(stream, fcw,  256, NQUES, wFc, 128, NQUESP);

  // 1) embeddings
  embed_kernel<<<EW_BLOCKS, 256, 0, stream>>>(q, qa, diff, qtab, qatab, QA, QD);

  // 2) GRU: input projections (WMMA), recurrence, output projection (WMMA)
  launch_gemm(stream, QA, DQ, wGz, bz, XG,        768, DQ, DQ, DQ, 0);
  launch_gemm(stream, QA, DQ, wGr, br, XG + DQ,   768, DQ, DQ, DQ, 0);
  launch_gemm(stream, QA, DQ, wGn, bn, XG + 2*DQ, 768, DQ, DQ, DQ, 0);
  gru_scan_kernel<<<BQ, 256, 0, stream>>>(XG, Uz, Ur, Un, HS);
  launch_gemm(stream, HS, DQ, wGo, bo, GRUO, DQ, DQ, DQ, DQ, 0);

  // 3) last Mamba layer on both streams (h = qa_emb, d = q_diff_emb)
  for (int s = 0; s < 2; ++s) {
    const float* X    = s ? QD   : QA;
    float*       OUTS = s ? OUTD : OUTH;
    for (int dir = 0; dir < 2; ++dir) {          // 0 = fwd, 1 = bwd (time-flipped)
      const int pb = LBASE + (dir ? 9 : 0);
      const float* conv_w = (const float*)d_in[pb+1];
      const float* conv_b = (const float*)d_in[pb+2];
      const float* b_dt   = (const float*)d_in[pb+5];
      const float* A_log  = (const float*)d_in[pb+6];
      const float* Dp     = (const float*)d_in[pb+7];

      launch_gemm(stream, X, DQ, wWin[dir], nullptr, BIG0, 1024, 1024, 1024, 256, 0);
      conv_silu_kernel<<<(MQ*DIQ)/256, 256, 0, stream>>>(BIG0, conv_w, conv_b, XH, dir);
      launch_gemm(stream, XH, DIQ, wWx[dir], nullptr, XDBL, 64, 64, 64, 512, 0);
      launch_gemm(stream, XDBL, 64, wWdt[dir], b_dt, DTb, DIQ, 512, 512, 32, 2); // softplus
      scan_kernel<<<2*BQ, 256, 0, stream>>>(XH, DTb, XDBL, A_log, Dp, BIG0, Yb, dir);
      launch_gemm(stream, Yb, DIQ, wWou[dir], nullptr, TMPA, DQ, 256, 256, 512, 0);
      bi_accum_kernel<<<EW_BLOCKS, 256, 0, stream>>>(TMPA, BI, dir, dir == 0 ? 1 : 0);
    }
    const float* ln_g = (const float*)d_in[LBASE+18];
    const float* ln_b = (const float*)d_in[LBASE+19];
    const float* b1   = (const float*)d_in[LBASE+21];
    const float* b2   = (const float*)d_in[LBASE+23];
    const float* fg   = (const float*)d_in[LBASE+24];
    const float* fb   = (const float*)d_in[LBASE+25];

    add_ln_kernel<<<MQ, 256, 0, stream>>>(BI, X, ln_g, ln_b, Hb);
    launch_gemm(stream, Hb, DQ, wF1, b1, BIG0, 1024, 1024, 1024, 256, 1);  // gelu
    launch_gemm(stream, BIG0, 1024, wF2, b2, TMPA, DQ, 256, 256, 1024, 0);
    add_ln_kernel<<<MQ, 256, 0, stream>>>(TMPA, Hb, fg, fb, OUTS);
  }

  // 4) mix + output head
  mix_kernel<<<EW_BLOCKS, 256, 0, stream>>>(OUTH, OUTD, GRUO, a1, a2, a3, TMPA);
  launch_gemm(stream, TMPA, DQ, wMix, mixb, BI, DQ, DQ, DQ, DQ, 0);
  launch_gemm(stream, BI, DQ, wFc, fcb, out, NQUES, NQUES, NQUESP, DQ, 0);
}
